// ManifoldTransformer_73615739454021
// MI455X (gfx1250) — matrix operand order repaired and checked by reading, not compiled, not run on hardware
//
#include <hip/hip_runtime.h>
#include <math.h>

// ---------------------------------------------------------------------------
// GPT-2 style transformer forward for MI455X (gfx1250), bf16 WMMA + f32 acc.
// ---------------------------------------------------------------------------

typedef __attribute__((ext_vector_type(16))) __bf16 v16bf;
typedef __attribute__((ext_vector_type(8)))  float  v8f;

#define V_   50257
#define D_   512
#define H_   8
#define DK_  64
#define DFF_ 2048
#define L_   4
#define B_   2
#define S_   1024
#define T_   (B_*S_)

#define ACT_NONE 0
#define ACT_GELU 1

#define WMMA_BF16(a, b, c) \
  __builtin_amdgcn_wmma_f32_16x16x32_bf16(false, (a), false, (b), (short)0, (c), false, false)

// convert 8 consecutive fp32 -> bf16 into fragment r[off..off+7]
__device__ __forceinline__ void cvt8(v16bf& r, int off, const float* __restrict__ p){
  float4 a = ((const float4*)p)[0];
  float4 b = ((const float4*)p)[1];
  r[off+0]=(__bf16)a.x; r[off+1]=(__bf16)a.y; r[off+2]=(__bf16)a.z; r[off+3]=(__bf16)a.w;
  r[off+4]=(__bf16)b.x; r[off+5]=(__bf16)b.y; r[off+6]=(__bf16)b.z; r[off+7]=(__bf16)b.w;
}

// A-operand fragment for K-step kb: k = kb+kh*16+0..15
__device__ __forceinline__ v16bf load_afrag(const float* __restrict__ row, int kb, int kh){
  v16bf r;
  cvt8(r, 0, row + kb + kh * 16);
  cvt8(r, 8, row + kb + kh * 16 + 8);
  return r;
}

// B-operand fragment: 16 consecutive k values per lane: kb + kh*16 + 0..15
__device__ __forceinline__ v16bf load_bfrag(const float* __restrict__ p){
  v16bf r;
  cvt8(r, 0, p);
  cvt8(r, 8, p + 8);
  return r;
}

__device__ __forceinline__ void store_tile(const v8f& acc, int row0, int col, int N,
                                           float alpha, const float* __restrict__ bias,
                                           int act, const float* __restrict__ R,
                                           float* __restrict__ C, int ldc)
{
  if (col >= N) return;
  const float bv = bias ? bias[col] : 0.f;
  #pragma unroll
  for (int j = 0; j < 8; j++) {
    const int row = row0 + j;
    float vv = acc[j] * alpha + bv;
    if (act == ACT_GELU) vv = 0.5f * vv * (1.0f + erff(vv * 0.70710678118654752f));
    if (R) vv += R[(size_t)row * ldc + col];
    C[(size_t)row * ldc + col] = vv;
  }
}

// ---------------------------------------------------------------------------
// Generic NT GEMM:  C[m,n] = act(alpha * sum_k A[m,k]*W[n,k] + bias[n]) + R[m,n]
// A: [M,K] lda row-major, W: [N,K] ldw row-major, batched via blockIdx.z.
// 256 threads = 8 waves arranged 4(M) x 2(N); each wave computes a 32x32 tile
// (2 A-frags x 2 B-frags -> 4 WMMAs per K-step).  Block tile = 128M x 64N.
// Requires M % 128 == 0 and K % 32 == 0 (true for all uses); N arbitrary.
// ---------------------------------------------------------------------------
__global__ __launch_bounds__(256)
void gemm_nt_bf16(const float* __restrict__ A, int lda,
                  const float* __restrict__ W, int ldw,
                  const float* __restrict__ bias,
                  const float* __restrict__ R,      // residual, layout == C
                  float* __restrict__ C, int ldc,
                  int M, int N, int K,
                  float alpha, int act, int Hh,
                  long long sAb, long long sAh,
                  long long sWb, long long sWh,
                  long long sCb, long long sCh)
{
  const int z  = blockIdx.z;
  const int zb = z / Hh, zh = z - zb * Hh;
  A += (long long)zb * sAb + (long long)zh * sAh;
  W += (long long)zb * sWb + (long long)zh * sWh;
  const long long coff = (long long)zb * sCb + (long long)zh * sCh;
  C += coff;
  if (R) R += coff;

  const int tid  = threadIdx.x;
  const int wave = tid >> 5;
  const int lane = tid & 31;
  const int bm = blockIdx.x * 128 + (wave >> 1) * 32;  // 32-row wave strip
  const int bn = blockIdx.y * 64  + (wave & 1) * 32;   // 32-col wave strip

  const int mr = lane & 15;     // row within A tile / col within B tile
  const int kh = lane >> 4;     // lane-half selects K sub-range

  const float* ar0 = A + (size_t)(bm + mr) * lda;
  const float* ar1 = A + (size_t)(bm + 16 + mr) * lda;
  const int c0 = bn + mr;
  const int c1 = c0 + 16;

  v8f acc00 = {}; v8f acc01 = {};
  v8f acc10 = {}; v8f acc11 = {};

  if (bn + 31 < N) {
    // ---- fast path: all 32 columns valid, branch-free inner loop ----
    const float* w0 = W + (size_t)c0 * ldw;
    const float* w1 = W + (size_t)c1 * ldw;
    for (int kb = 0; kb < K; kb += 32) {
      __builtin_prefetch(ar0 + kb + 128, 0, 3);
      __builtin_prefetch(w0 + kb + 128, 0, 3);
      const v16bf a0 = load_afrag(ar0, kb, kh);
      const v16bf a1 = load_afrag(ar1, kb, kh);
      const int kw = kb + kh * 16;
      const v16bf b0 = load_bfrag(w0 + kw);
      const v16bf b1 = load_bfrag(w1 + kw);
      acc00 = WMMA_BF16(a0, b0, acc00);
      acc01 = WMMA_BF16(a0, b1, acc01);
      acc10 = WMMA_BF16(a1, b0, acc10);
      acc11 = WMMA_BF16(a1, b1, acc11);
    }
  } else {
    // ---- boundary path (vocab-head edge): per-lane column guards ----
    const bool v0 = (c0 < N), v1 = (c1 < N);
    const float* w0 = W + (size_t)(v0 ? c0 : 0) * ldw;
    const float* w1 = W + (size_t)(v1 ? c1 : 0) * ldw;
    const v16bf zf = {};
    for (int kb = 0; kb < K; kb += 32) {
      const v16bf a0 = load_afrag(ar0, kb, kh);
      const v16bf a1 = load_afrag(ar1, kb, kh);
      const int kw = kb + kh * 16;
      v16bf b0 = zf, b1 = zf;
      if (v0) b0 = load_bfrag(w0 + kw);
      if (v1) b1 = load_bfrag(w1 + kw);
      acc00 = WMMA_BF16(a0, b0, acc00);
      acc01 = WMMA_BF16(a0, b1, acc01);
      acc10 = WMMA_BF16(a1, b0, acc10);
      acc11 = WMMA_BF16(a1, b1, acc11);
    }
  }

  // C layout: lane n = lane&15 holds column, rows (lane>>4)*8 + j
  const int n  = lane & 15;
  const int mh = lane >> 4;
  store_tile(acc00, bm + mh * 8,      bn + n,      N, alpha, bias, act, R, C, ldc);
  store_tile(acc01, bm + mh * 8,      bn + 16 + n, N, alpha, bias, act, R, C, ldc);
  store_tile(acc10, bm + 16 + mh * 8, bn + n,      N, alpha, bias, act, R, C, ldc);
  store_tile(acc11, bm + 16 + mh * 8, bn + 16 + n, N, alpha, bias, act, R, C, ldc);
}

// ---------------------------------------------------------------------------
// token + positional embedding:  x[t,d] = emb[tok[t],d] + pos[t%S,d]
// ---------------------------------------------------------------------------
__global__ __launch_bounds__(256)
void embed_kernel(const int* __restrict__ tok, const float* __restrict__ emb,
                  const float* __restrict__ pos, float* __restrict__ x)
{
  const int idx = blockIdx.x * 256 + threadIdx.x;
  if (idx >= T_ * D_) return;
  const int d = idx % D_;
  const int t = idx / D_;
  const int s = t % S_;
  x[idx] = emb[(size_t)tok[t] * D_ + d] + pos[(size_t)s * D_ + d];
}

// ---------------------------------------------------------------------------
// LayerNorm per row (D=512), 256 threads per row
// ---------------------------------------------------------------------------
__global__ __launch_bounds__(256)
void ln_kernel(const float* __restrict__ xin, const float* __restrict__ g,
               const float* __restrict__ bt, float* __restrict__ out)
{
  const int row = blockIdx.x;
  const int tid = threadIdx.x;
  const float* xr = xin + (size_t)row * D_;

  float s = 0.f, s2 = 0.f;
  for (int d = tid; d < D_; d += 256) { float v = xr[d]; s += v; s2 += v * v; }

  __shared__ float sh1[256], sh2[256];
  sh1[tid] = s; sh2[tid] = s2;
  __syncthreads();
  for (int off = 128; off > 0; off >>= 1) {
    if (tid < off) { sh1[tid] += sh1[tid + off]; sh2[tid] += sh2[tid + off]; }
    __syncthreads();
  }
  const float mean = sh1[0] * (1.0f / D_);
  const float var  = sh2[0] * (1.0f / D_) - mean * mean;
  const float rs   = rsqrtf(var + 1e-5f);

  for (int d = tid; d < D_; d += 256)
    out[(size_t)row * D_ + d] = (xr[d] - mean) * rs * g[d] + bt[d];
}

// ---------------------------------------------------------------------------
// causal softmax in place over attn rows of length S; block per (b,h,i) row
// ---------------------------------------------------------------------------
__global__ __launch_bounds__(256)
void softmax_kernel(float* __restrict__ attn)
{
  const int r = blockIdx.x;
  const int i = r & (S_ - 1);
  float* row = attn + (size_t)r * S_;
  const int tid = threadIdx.x;

  float lv[4];
  float mx = -1e30f;
  #pragma unroll
  for (int jj = 0; jj < 4; jj++) {
    const int j = jj * 256 + tid;
    const float val = (j <= i) ? row[j] : -1e30f;
    lv[jj] = val;
    mx = fmaxf(mx, val);
  }

  __shared__ float sh[256];
  sh[tid] = mx; __syncthreads();
  for (int off = 128; off > 0; off >>= 1) {
    if (tid < off) sh[tid] = fmaxf(sh[tid], sh[tid + off]);
    __syncthreads();
  }
  const float m = sh[0];
  __syncthreads();

  float sm = 0.f;
  #pragma unroll
  for (int jj = 0; jj < 4; jj++) {
    const int j = jj * 256 + tid;
    const float e = (j <= i) ? expf(lv[jj] - m) : 0.f;
    lv[jj] = e;
    sm += e;
  }
  sh[tid] = sm; __syncthreads();
  for (int off = 128; off > 0; off >>= 1) {
    if (tid < off) sh[tid] += sh[tid + off];
    __syncthreads();
  }
  const float inv = 1.0f / sh[0];

  #pragma unroll
  for (int jj = 0; jj < 4; jj++) row[jj * 256 + tid] = lv[jj] * inv;
}

// ---------------------------------------------------------------------------
// vT[(b*H+h)*DK + d][t] = v[b*S+t][h*DK + d]   (makes attn@v an NT gemm)
// ---------------------------------------------------------------------------
__global__ __launch_bounds__(256)
void transpose_v_kernel(const float* __restrict__ v, float* __restrict__ vT)
{
  const int idx = blockIdx.x * 256 + threadIdx.x;
  if (idx >= T_ * D_) return;
  const int t  = idx & (S_ - 1);
  const int r1 = idx >> 10;          // / S_
  const int d  = r1 & (DK_ - 1);
  const int r2 = r1 >> 6;            // / DK_
  const int hh = r2 & (H_ - 1);
  const int b  = r2 >> 3;            // / H_
  vT[idx] = v[((size_t)(b * S_ + t)) * D_ + hh * DK_ + d];
}

// ---------------------------------------------------------------------------
extern "C" void kernel_launch(void* const* d_in, const int* in_sizes, int n_in,
                              void* d_out, int out_size, void* d_ws, size_t ws_size,
                              hipStream_t stream)
{
  (void)in_sizes; (void)n_in; (void)out_size; (void)ws_size;

  const int*   tok  = (const int*)  d_in[0];
  const float* emb  = (const float*)d_in[1];
  const float* pos  = (const float*)d_in[2];
  const float* Wq   = (const float*)d_in[3];  const float* bq = (const float*)d_in[4];
  const float* Wk   = (const float*)d_in[5];  const float* bk = (const float*)d_in[6];
  const float* Wv   = (const float*)d_in[7];  const float* bv = (const float*)d_in[8];
  const float* Wo   = (const float*)d_in[9];  const float* bo = (const float*)d_in[10];
  const float* ln1g = (const float*)d_in[11]; const float* ln1b = (const float*)d_in[12];
  const float* W1   = (const float*)d_in[13]; const float* b1 = (const float*)d_in[14];
  const float* W2   = (const float*)d_in[15]; const float* b2 = (const float*)d_in[16];
  const float* ln2g = (const float*)d_in[17]; const float* ln2b = (const float*)d_in[18];
  const float* lnfg = (const float*)d_in[19]; const float* lnfb = (const float*)d_in[20];
  const float* Wh   = (const float*)d_in[21]; const float* bh = (const float*)d_in[22];
  float* out = (float*)d_out;

  float* ws = (float*)d_ws;
  const size_t TD = (size_t)T_ * D_;
  float* x    = ws;            // [T,D]
  float* h    = ws + 1 * TD;   // [T,D]  LN output
  float* q    = ws + 2 * TD;   // [T,D]
  float* k    = ws + 3 * TD;   // [T,D]
  float* v    = ws + 4 * TD;   // [T,D]
  float* o    = ws + 5 * TD;   // [T,D]
  float* vT   = ws + 6 * TD;   // [B,H,DK,S]
  float* hf   = ws + 7 * TD;   // [T,DFF] (4*TD)
  float* attn = ws + 11 * TD;  // [B,H,S,S] (16*TD)

  const dim3 blk(256);

  auto GEMM = [&](const float* Ap, int lda, const float* Wp, int ldw,
                  const float* biasp, const float* Rp, float* Cp, int ldc,
                  int M, int N, int K, float alpha, int act,
                  int Z, int Hh,
                  long long sAb, long long sAh,
                  long long sWb, long long sWh,
                  long long sCb, long long sCh) {
    dim3 grid((unsigned)(M / 128), (unsigned)((N + 63) / 64), (unsigned)Z);
    gemm_nt_bf16<<<grid, blk, 0, stream>>>(Ap, lda, Wp, ldw, biasp, Rp, Cp, ldc,
                                           M, N, K, alpha, act, Hh,
                                           sAb, sAh, sWb, sWh, sCb, sCh);
  };

  embed_kernel<<<(T_ * D_ + 255) / 256, blk, 0, stream>>>(tok, emb, pos, x);

  for (int l = 0; l < L_; l++) {
    const size_t wDD = (size_t)l * D_ * D_;
    const size_t wFD = (size_t)l * DFF_ * D_;

    // ---- attention ----
    ln_kernel<<<T_, blk, 0, stream>>>(x, ln1g + l * D_, ln1b + l * D_, h);
    GEMM(h, D_, Wq + wDD, D_, bq + l * D_, nullptr, q, D_, T_, D_, D_, 1.f, ACT_NONE,
         1, 1, 0, 0, 0, 0, 0, 0);
    GEMM(h, D_, Wk + wDD, D_, bk + l * D_, nullptr, k, D_, T_, D_, D_, 1.f, ACT_NONE,
         1, 1, 0, 0, 0, 0, 0, 0);
    GEMM(h, D_, Wv + wDD, D_, bv + l * D_, nullptr, v, D_, T_, D_, D_, 1.f, ACT_NONE,
         1, 1, 0, 0, 0, 0, 0, 0);

    // scores[b,h] = (q_bh @ k_bh^T) / sqrt(dk)
    GEMM(q, D_, k, D_, nullptr, nullptr, attn, S_, S_, S_, DK_, 0.125f, ACT_NONE,
         B_ * H_, H_,
         (long long)S_ * D_, (long long)DK_,
         (long long)S_ * D_, (long long)DK_,
         (long long)H_ * S_ * S_, (long long)S_ * S_);

    softmax_kernel<<<B_ * H_ * S_, blk, 0, stream>>>(attn);
    transpose_v_kernel<<<(T_ * D_ + 255) / 256, blk, 0, stream>>>(v, vT);

    // o[b,:,h,:] = attn[b,h] @ v[b,:,h,:]   (W operand = vT, NT form)
    GEMM(attn, S_, vT, S_, nullptr, nullptr, o, D_, S_, DK_, S_, 1.f, ACT_NONE,
         B_ * H_, H_,
         (long long)H_ * S_ * S_, (long long)S_ * S_,
         (long long)H_ * DK_ * S_, (long long)DK_ * S_,
         (long long)S_ * D_, (long long)DK_);

    // x = x + o @ Wo^T + bo   (in-place residual)
    GEMM(o, D_, Wo + wDD, D_, bo + l * D_, x, x, D_, T_, D_, D_, 1.f, ACT_NONE,
         1, 1, 0, 0, 0, 0, 0, 0);

    // ---- FFN ----
    ln_kernel<<<T_, blk, 0, stream>>>(x, ln2g + l * D_, ln2b + l * D_, h);
    GEMM(h, D_, W1 + wFD, D_, b1 + l * DFF_, nullptr, hf, DFF_, T_, DFF_, D_,
         1.f, ACT_GELU, 1, 1, 0, 0, 0, 0, 0, 0);
    GEMM(hf, DFF_, W2 + wFD, DFF_, b2 + l * D_, x, x, D_, T_, D_, DFF_,
         1.f, ACT_NONE, 1, 1, 0, 0, 0, 0, 0, 0);
  }

  // final LN + vocab head -> logits [T, V] fp32
  ln_kernel<<<T_, blk, 0, stream>>>(x, lnfg, lnfb, h);
  GEMM(h, D_, Wh, D_, bh, nullptr, out, V_, T_, V_, D_, 1.f, ACT_NONE,
       1, 1, 0, 0, 0, 0, 0, 0);
}
